// VectorQuantizerEMA_70351564308893
// MI455X (gfx1250) — compile-verified, matrix operand order checked by reading, not run on hardware
//
#include <hip/hip_runtime.h>

typedef __attribute__((ext_vector_type(16))) __bf16 v16bf;
typedef __attribute__((ext_vector_type(8)))  __bf16 v8bf;
typedef __attribute__((ext_vector_type(8)))  float  v8f;
typedef __attribute__((ext_vector_type(4)))  float  v4f;
typedef int v4i_g __attribute__((vector_size(16)));   // matches builtin param type

#define KCB   8192
#define DIM   256
#define NTOK  16384
#define BETA_V  0.1f
#define DECAY_V 0.99f
#define OMD_V   0.01f
#define EPS_V   1e-5f

// LDS tile geometry: 32 codebook rows per tile, rows padded to 264 bf16 (528 B)
// -> ds_load_b128 from the 16 lanes hits banks 0,4,8,...,60: conflict-free.
#define TROWS   32
#define LPITCH  264
#define NTILES  (KCB / TROWS)

// output offsets (floats), in reference return order
#define ZQ_OFF    ((size_t)0)          // 8*2048*256
#define CODES_OFF ((size_t)4194304)    // 16384
#define LOSS_OFF  ((size_t)4210688)    // 1
#define NCB_OFF   ((size_t)4210689)    // 8192*256 new_codebook
#define NCS_OFF   ((size_t)6307841)    // 8192     new_cluster_size
#define NEW_OFF   ((size_t)6316033)    // 8192*256 new_ema_w

#if __has_builtin(__builtin_amdgcn_global_load_async_to_lds_b128)
#define HAS_ASYNC_LDS 1
#else
#define HAS_ASYNC_LDS 0
#endif

__device__ __forceinline__ void wait_async_lds() {
#if HAS_ASYNC_LDS
#if __has_builtin(__builtin_amdgcn_s_wait_asynccnt)
    __builtin_amdgcn_s_wait_asynccnt(0);
#else
    asm volatile("s_wait_asynccnt 0x0" ::: "memory");
#endif
#endif
}

// ---------------- prep: ||c||^2 and bf16 codebook copy -------------------
__global__ __launch_bounds__(256)
void vq_prep(const float* __restrict__ cb, float* __restrict__ cnorm,
             __bf16* __restrict__ cb16)
{
    int wid  = (blockIdx.x * blockDim.x + threadIdx.x) >> 5;
    int lane = threadIdx.x & 31;
    if (wid >= KCB) return;
    const float* row = cb + (size_t)wid * DIM;
    __bf16* orow = cb16 + (size_t)wid * DIM;
    float s = 0.f;
#pragma unroll
    for (int i = 0; i < 8; ++i) {
        int d = lane * 8 + i;
        float v = row[d];
        s += v * v;
        orow[d] = (__bf16)v;
    }
#pragma unroll
    for (int m = 16; m >= 1; m >>= 1) s += __shfl_xor(s, m, 32);
    if (lane == 0) cnorm[wid] = s;
}

// ---------------- main: WMMA distance GEMM + argmin + gather/scatter -----
__global__ __launch_bounds__(256)
void vq_main(const float* __restrict__ z_e, const float* __restrict__ cb,
             const __bf16* __restrict__ cb16, const float* __restrict__ cnorm,
             float* __restrict__ dw, float* __restrict__ n_i,
             float* __restrict__ loss, float* __restrict__ out)
{
    __shared__ __bf16 sB[2][TROWS * LPITCH + 8];
    __shared__ float  sCn[2][TROWS];

    int wave = threadIdx.x >> 5;                      // 0..7
    int lane = threadIdx.x & 31;
    int l15  = lane & 15;
    int half = lane >> 4;
    int m0   = blockIdx.x * 128 + wave * 16;          // 16 rows per wave

    // ---- A fragments: 16 rows x 256 K, bf16, register-resident ----
    // ISA 16-bit A layout: lane L<16 row L holds K {kb..kb+7, kb+16..kb+23};
    // lane L+16 same row holds {kb+8..kb+15, kb+24..kb+31}.
    const float* zrow = z_e + (size_t)(m0 + l15) * DIM;
    v16bf A[8];
#pragma unroll
    for (int f = 0; f < 8; ++f) {
        int kb = f * 32 + half * 8;
        v4f x0 = *(const v4f*)(zrow + kb);
        v4f x1 = *(const v4f*)(zrow + kb + 4);
        v4f x2 = *(const v4f*)(zrow + kb + 16);
        v4f x3 = *(const v4f*)(zrow + kb + 20);
        v16bf a;
#pragma unroll
        for (int i = 0; i < 4; ++i) {
            a[i]      = (__bf16)x0[i];
            a[4 + i]  = (__bf16)x1[i];
            a[8 + i]  = (__bf16)x2[i];
            a[12 + i] = (__bf16)x3[i];
        }
        A[f] = a;
    }

    float best[8];
    int   bidx[8];
#pragma unroll
    for (int j = 0; j < 8; ++j) { best[j] = 3.4e38f; bidx[j] = 0x7fffffff; }

    // ---- tile stager: 32 rows x 512B, 256 threads x 64B each ----
    // thread t: row = t>>3, 64B chunk = t&7
    auto stage_tile = [&](int buf, int n0) {
        int row   = threadIdx.x >> 3;
        int chunk = threadIdx.x & 7;
        const __bf16* gsrc = cb16 + (size_t)(n0 + row) * DIM + chunk * 32;
        __bf16* ldst = &sB[buf][row * LPITCH + chunk * 32];
#if HAS_ASYNC_LDS
#pragma unroll
        for (int q = 0; q < 4; ++q) {
            v4i_g* lg = (v4i_g*)(ldst + q * 8);
            __builtin_amdgcn_global_load_async_to_lds_b128(
                (__attribute__((address_space(1))) v4i_g*)(uintptr_t)(gsrc + q * 8),
                (__attribute__((address_space(3))) v4i_g*)lg,
                0, 0);
        }
#else
#pragma unroll
        for (int q = 0; q < 4; ++q)
            ((uint4*)ldst)[q] = ((const uint4*)gsrc)[q];
#endif
        if (threadIdx.x < TROWS) sCn[buf][threadIdx.x] = cnorm[n0 + threadIdx.x];
    };

    stage_tile(0, 0);
    wait_async_lds();
    __syncthreads();

    for (int t = 0; t < NTILES; ++t) {
        int cur = t & 1;
        if (t + 1 < NTILES) stage_tile(cur ^ 1, (t + 1) * TROWS);   // overlap DMA

        // two independent WMMA accumulator chains (N-tiles n0..+15, n0+16..+31)
        const __bf16* r0 = &sB[cur][(size_t)l15 * LPITCH];
        const __bf16* r1 = &sB[cur][(size_t)(l15 + 16) * LPITCH];
        v8f acc0 = {}, acc1 = {};
#pragma unroll
        for (int f = 0; f < 8; ++f) {
            int kb = f * 32 + half * 8;
            v8bf lo0 = *(const v8bf*)(r0 + kb);
            v8bf hi0 = *(const v8bf*)(r0 + kb + 16);
            v8bf lo1 = *(const v8bf*)(r1 + kb);
            v8bf hi1 = *(const v8bf*)(r1 + kb + 16);
            v16bf b0, b1;
#pragma unroll
            for (int i = 0; i < 8; ++i) {
                b0[i] = lo0[i]; b0[8 + i] = hi0[i];
                b1[i] = lo1[i]; b1[8 + i] = hi1[i];
            }
            acc0 = __builtin_amdgcn_wmma_f32_16x16x32_bf16(
                false, A[f], false, b0, (short)0, acc0, false, false);
            acc1 = __builtin_amdgcn_wmma_f32_16x16x32_bf16(
                false, A[f], false, b1, (short)0, acc1, false, false);
        }

        // argmin metric: ||c||^2 - 2 z.c   (||z||^2 row-constant, dropped)
        int   n0  = t * TROWS;
        float cn0 = sCn[cur][l15];
        float cn1 = sCn[cur][l15 + 16];
        int   na  = n0 + l15;
        int   nb  = n0 + 16 + l15;
#pragma unroll
        for (int j = 0; j < 8; ++j) {
            float d0 = cn0 - 2.0f * acc0[j];
            if (d0 < best[j]) { best[j] = d0; bidx[j] = na; }
            float d1 = cn1 - 2.0f * acc1[j];
            if (d1 < best[j]) { best[j] = d1; bidx[j] = nb; }
        }

        wait_async_lds();
        __syncthreads();
    }

    // butterfly argmin across the 16 lanes of each C-layout half,
    // first-index tie-break to match jnp.argmin
#pragma unroll
    for (int m = 8; m >= 1; m >>= 1) {
#pragma unroll
        for (int j = 0; j < 8; ++j) {
            float ob = __shfl_xor(best[j], m, 32);
            int   oi = __shfl_xor(bidx[j], m, 32);
            if (ob < best[j] || (ob == best[j] && oi < bidx[j])) {
                best[j] = ob; bidx[j] = oi;
            }
        }
    }

    // codes (as float): lanes 0..15 hold rows m0..m0+7, lanes 16..31 rows +8
    if (lane == 0) {
#pragma unroll
        for (int j = 0; j < 8; ++j) out[CODES_OFF + m0 + j] = (float)bidx[j];
    } else if (lane == 16) {
#pragma unroll
        for (int j = 0; j < 8; ++j) out[CODES_OFF + m0 + 8 + j] = (float)bidx[j];
    }

    // gather z_q, loss, scatter EMA stats (fp32 atomics -> L2)
    float lsum = 0.f;
#pragma unroll
    for (int m = 0; m < 16; ++m) {
        int code = __shfl(bidx[m & 7], (m < 8) ? 0 : 16, 32);
        const float* zr = z_e + (size_t)(m0 + m) * DIM;
        const float* cr = cb  + (size_t)code * DIM;
        float* qo = out + ZQ_OFF + (size_t)(m0 + m) * DIM;
        float* dr = dw + (size_t)code * DIM;
#pragma unroll
        for (int i = 0; i < 8; ++i) {
            int d = lane * 8 + i;
            float z = zr[d];
            float c = cr[d];
            qo[d] = c;                       // z_q_st forward value == z_q
            float e = z - c;
            lsum += e * e;
            atomicAdd(&dr[d], z);            // dw segment-sum
        }
        if (lane == 0) atomicAdd(&n_i[code], 1.0f);
    }
#pragma unroll
    for (int m = 16; m >= 1; m >>= 1) lsum += __shfl_xor(lsum, m, 32);
    if (lane == 0) atomicAdd(loss, lsum);
}

// ---------------- EMA update + final loss ---------------------------------
__global__ __launch_bounds__(256)
void vq_ema(const float* __restrict__ ema_cs, const float* __restrict__ ema_w,
            const float* __restrict__ dw, const float* __restrict__ n_i,
            const float* __restrict__ loss, float* __restrict__ out)
{
    int wid  = (blockIdx.x * blockDim.x + threadIdx.x) >> 5;
    int lane = threadIdx.x & 31;
    if (blockIdx.x == 0 && threadIdx.x == 0)
        out[LOSS_OFF] = BETA_V * loss[0] / (float)((size_t)NTOK * DIM);
    if (wid >= KCB) return;
    float ncs = DECAY_V * ema_cs[wid] + OMD_V * n_i[wid];
    if (lane == 0) out[NCS_OFF + wid] = ncs;
    float inv = 1.0f / (ncs + EPS_V);
    const float* wr = ema_w + (size_t)wid * DIM;
    const float* dr = dw    + (size_t)wid * DIM;
#pragma unroll
    for (int i = 0; i < 8; ++i) {
        int d = lane * 8 + i;
        float nw = DECAY_V * wr[d] + OMD_V * dr[d];
        out[NEW_OFF + (size_t)wid * DIM + d] = nw;
        out[NCB_OFF + (size_t)wid * DIM + d] = nw * inv;
    }
}

extern "C" void kernel_launch(void* const* d_in, const int* in_sizes, int n_in,
                              void* d_out, int out_size, void* d_ws, size_t ws_size,
                              hipStream_t stream)
{
    (void)in_sizes; (void)n_in; (void)out_size; (void)ws_size;
    const float* z_e = (const float*)d_in[0];
    const float* cb  = (const float*)d_in[1];
    const float* ecs = (const float*)d_in[2];
    const float* ew  = (const float*)d_in[3];
    float* out = (float*)d_out;

    // workspace layout (bytes): [cb16 bf16 4MB][cnorm 32KB][dw 8MB][n_i 32KB][loss 16B]
    char* ws = (char*)d_ws;
    __bf16* cb16 = (__bf16*)ws;
    float* cnorm = (float*)(ws + (size_t)KCB * DIM * 2);
    float* dw    = (float*)(ws + (size_t)KCB * DIM * 2 + (size_t)KCB * 4);
    float* n_i   = dw + (size_t)KCB * DIM;
    float* loss  = n_i + KCB;

    // zero only the accumulator region (graph-capture-safe)
    (void)hipMemsetAsync(dw, 0, ((size_t)KCB * DIM + KCB + 4) * sizeof(float), stream);

    vq_prep<<<KCB / 8, 256, 0, stream>>>(cb, cnorm, cb16);
    vq_main<<<NTOK / 128, 256, 0, stream>>>(z_e, cb, cb16, cnorm, dw, n_i, loss, out);
    vq_ema<<<KCB / 8, 256, 0, stream>>>(ecs, ew, dw, n_i, loss, out);
}
